// RetrievalMemory_52810917871830
// MI455X (gfx1250) — matrix operand order repaired and checked by reading, not compiled, not run on hardware
//
#include <hip/hip_runtime.h>

typedef __attribute__((ext_vector_type(8)))  float  v8f;
typedef __attribute__((ext_vector_type(8)))  __bf16 v8bf;
typedef __attribute__((ext_vector_type(16))) __bf16 v16bf;

__device__ __forceinline__ __bf16 f2bf(float f) {
  unsigned u = __builtin_bit_cast(unsigned, f);
  u += 0x7FFFu + ((u >> 16) & 1u);           // round-to-nearest-even
  unsigned short h = (unsigned short)(u >> 16);
  return __builtin_bit_cast(__bf16, h);
}

__device__ __forceinline__ v16bf cat8(v8bf lo, v8bf hi) {
  v16bf r;
#pragma unroll
  for (int i = 0; i < 8; ++i) { r[i] = lo[i]; r[i + 8] = hi[i]; }
  return r;
}

__device__ __forceinline__ v8f wmma_bf16(v16bf a, v16bf b, v8f c) {
  return __builtin_amdgcn_wmma_f32_16x16x32_bf16(false, a, false, b,
                                                 (short)0, c, false, false);
}

// ---------------- elementwise f32 -> bf16 ----------------
__global__ void f32_to_bf16_kernel(const float* __restrict__ in,
                                   __bf16* __restrict__ out, long long n) {
  long long i = (long long)blockIdx.x * blockDim.x + threadIdx.x;
  if (i < n) out[i] = f2bf(in[i]);
}

// ---------------- mean-pool x -> slots (bf16) ----------------
__global__ void pool_slots_kernel(const float* __restrict__ x,
                                  __bf16* __restrict__ slots_b,
                                  int S, int C, int R, int T, long long n) {
  long long i = (long long)blockIdx.x * blockDim.x + threadIdx.x;  // over B*S*C
  if (i >= n) return;
  int c = (int)(i % C);
  long long bs = i / C;
  int s = (int)(bs % S);
  int b = (int)(bs / S);
  const float* xp = x + ((long long)b * T + (long long)s * R) * C + c;
  float acc = 0.f;
  for (int r = 0; r < R; ++r) acc += xp[(long long)r * C];
  slots_b[i] = f2bf(acc * (1.0f / (float)R));
}

// ---------------- batched bf16 WMMA GEMM ----------------
// D[b][m][n] = alpha * sum_k A[b][m][k] * Bw[b][n][k]
// one wave computes a 32(M) x 64(N) tile, register ping-pong pipelined over K.
// Uniform-base + 32-bit lane-offset addressing (saddr+voffset form).
// grid = (N/64, M/32, nbatch); K must be a multiple of 64.
__global__ void __launch_bounds__(32)
gemm_bf16_wmma(const __bf16* __restrict__ A, const __bf16* __restrict__ Bw,
               float* __restrict__ Df, __bf16* __restrict__ Db,
               int N, int K,
               long long sA, long long sB, long long sD, float alpha) {
  const int lane = threadIdx.x;
  const int half = lane >> 4;         // lane group 0/1
  const int l16  = lane & 15;
  const int m0 = blockIdx.y << 5;
  const int n0 = blockIdx.x << 6;
  const int bz = blockIdx.z;

  const __bf16* __restrict__ Abase = A  + (long long)bz * sA;   // uniform
  const __bf16* __restrict__ Bbase = Bw + (long long)bz * sB;   // uniform

  // loop-invariant per-lane element offsets (32-bit, stay resident in VGPRs)
  // A operand rows m.. : lanes 0-15 hold K {k..k+7, k+16..k+23};
  //                      lanes 16-31 hold K {k+8..k+15, k+24..k+31}
  const int aoff0 = (m0 + l16) * K + (half << 3);
  const int aoff1 = aoff0 + 16 * K;
  // B operand: lane group 0 -> K {k..k+7, k+16..k+23}; lane group 1 -> K {k+8..k+15, k+24..k+31}
  const int boff0 = (n0 + l16) * K + (half << 3);
  const int boff1 = boff0 + 16 * K;
  const int boff2 = boff0 + 32 * K;
  const int boff3 = boff0 + 48 * K;

  auto ldAv = [&](const __bf16* ub, int off) -> v16bf {
    return cat8(*(const v8bf*)(ub + off), *(const v8bf*)(ub + off + 16));
  };
  auto ldBv = [&](const __bf16* ub, int off) -> v16bf {
    return cat8(*(const v8bf*)(ub + off), *(const v8bf*)(ub + off + 16));
  };

  v8f acc[8] = {};

  // prologue: set X holds K-step 0
  v16bf aX0 = ldAv(Abase, aoff0), aX1 = ldAv(Abase, aoff1);
  v16bf bX0 = ldBv(Bbase, boff0), bX1 = ldBv(Bbase, boff1);
  v16bf bX2 = ldBv(Bbase, boff2), bX3 = ldBv(Bbase, boff3);

  for (int k = 0; k < K; k += 64) {
    // issue loads for k+32 (always in-bounds: K % 64 == 0); uniform base advance
    const __bf16* Ay = Abase + k + 32;
    const __bf16* By = Bbase + k + 32;
    v16bf aY0 = ldAv(Ay, aoff0), aY1 = ldAv(Ay, aoff1);
    v16bf bY0 = ldBv(By, boff0), bY1 = ldBv(By, boff1);
    v16bf bY2 = ldBv(By, boff2), bY3 = ldBv(By, boff3);
    // compute set X
    acc[0] = wmma_bf16(aX0, bX0, acc[0]);
    acc[1] = wmma_bf16(aX0, bX1, acc[1]);
    acc[2] = wmma_bf16(aX0, bX2, acc[2]);
    acc[3] = wmma_bf16(aX0, bX3, acc[3]);
    acc[4] = wmma_bf16(aX1, bX0, acc[4]);
    acc[5] = wmma_bf16(aX1, bX1, acc[5]);
    acc[6] = wmma_bf16(aX1, bX2, acc[6]);
    acc[7] = wmma_bf16(aX1, bX3, acc[7]);
    // issue loads for k+64 (last iteration over-reads <=64B; ws is padded)
    const __bf16* Ax = Abase + k + 64;
    const __bf16* Bx = Bbase + k + 64;
    aX0 = ldAv(Ax, aoff0); aX1 = ldAv(Ax, aoff1);
    bX0 = ldBv(Bx, boff0); bX1 = ldBv(Bx, boff1);
    bX2 = ldBv(Bx, boff2); bX3 = ldBv(Bx, boff3);
    // compute set Y
    acc[0] = wmma_bf16(aY0, bY0, acc[0]);
    acc[1] = wmma_bf16(aY0, bY1, acc[1]);
    acc[2] = wmma_bf16(aY0, bY2, acc[2]);
    acc[3] = wmma_bf16(aY0, bY3, acc[3]);
    acc[4] = wmma_bf16(aY1, bY0, acc[4]);
    acc[5] = wmma_bf16(aY1, bY1, acc[5]);
    acc[6] = wmma_bf16(aY1, bY2, acc[6]);
    acc[7] = wmma_bf16(aY1, bY3, acc[7]);
  }

  // C/D layout: lanes 0-15: N=lane, VGPR r -> M=r ; lanes 16-31: M=r+8
  const long long dbase = (long long)bz * sD;
#pragma unroll
  for (int mi = 0; mi < 2; ++mi) {
#pragma unroll
    for (int j = 0; j < 4; ++j) {
      const int n = n0 + (j << 4) + l16;
#pragma unroll
      for (int r = 0; r < 8; ++r) {
        const int m = m0 + (mi << 4) + (half << 3) + r;
        const long long off = dbase + (long long)m * N + n;
        const float val = acc[mi * 4 + j][r] * alpha;
        if (Df) Df[off] = val;
        else    Db[off] = f2bf(val);
      }
    }
  }
}

// ---------------- top-8 + softmax + weighted gather of v ----------------
// one 256-thread block per (b,t); S == 256 scores scanned in-register
__global__ void __launch_bounds__(256)
topk_gather_kernel(const float* __restrict__ sc, const float* __restrict__ V,
                   __bf16* __restrict__ ret, int T, int S, int C) {
  const int bt  = blockIdx.x;
  const int b   = bt / T;
  const int tid = threadIdx.x;
  __shared__ float rv[256];
  __shared__ int   ri[256];
  __shared__ float topv[8];
  __shared__ int   topi[8];

  float val = sc[(long long)bt * S + tid];
  int   idx = tid;
#pragma unroll 1
  for (int pass = 0; pass < 8; ++pass) {
    rv[tid] = val; ri[tid] = idx;
    __syncthreads();
    for (int o = 128; o > 0; o >>= 1) {
      if (tid < o && rv[tid + o] > rv[tid]) { rv[tid] = rv[tid + o]; ri[tid] = ri[tid + o]; }
      __syncthreads();
    }
    if (tid == 0) { topv[pass] = rv[0]; topi[pass] = ri[0]; }
    __syncthreads();
    if (idx == topi[pass]) val = -3.402823466e38f;   // mask winner
    __syncthreads();
  }

  float w[8]; float sum = 0.f;
  const float mx = topv[0];
#pragma unroll
  for (int kk = 0; kk < 8; ++kk) { w[kk] = __expf(topv[kk] - mx); sum += w[kk]; }
  const float inv = 1.0f / sum;
  const float* Vb = V + (long long)b * S * C;
  for (int c = tid; c < C; c += 256) {
    float acc = 0.f;
#pragma unroll
    for (int kk = 0; kk < 8; ++kk)
      acc += w[kk] * Vb[(long long)topi[kk] * C + c];
    ret[(long long)bt * C + c] = f2bf(acc * inv);
  }
}

extern "C" void kernel_launch(void* const* d_in, const int* in_sizes, int n_in,
                              void* d_out, int out_size, void* d_ws, size_t ws_size,
                              hipStream_t stream) {
  (void)in_sizes; (void)n_in; (void)out_size;
  constexpr int B = 8, T = 2048, C = 1024, S = 256;
  const float* x  = (const float*)d_in[0];
  const float* Wq = (const float*)d_in[1];
  const float* Wk = (const float*)d_in[2];
  const float* Wv = (const float*)d_in[3];
  const float* Wp = (const float*)d_in[4];
  float* out = (float*)d_out;

  char* ws = (char*)d_ws;
  size_t off = 0;
  auto take = [&](size_t bytes) -> char* {
    char* p = ws + off;
    off += (bytes + 255) & ~(size_t)255;
    return p;
  };

  __bf16* xb  = (__bf16*)take((size_t)B * T * C * 2);   // also reused as `retrieved`
  __bf16* wqb = (__bf16*)take((size_t)C * C * 2);
  __bf16* wkb = (__bf16*)take((size_t)C * C * 2);
  __bf16* wvb = (__bf16*)take((size_t)C * C * 2);
  __bf16* wpb = (__bf16*)take((size_t)C * C * 2);
  __bf16* slb = (__bf16*)take((size_t)B * S * C * 2);
  __bf16* qb  = (__bf16*)take((size_t)B * T * C * 2);
  __bf16* kb  = (__bf16*)take((size_t)B * S * C * 2);
  float*  vf  = (float*) take((size_t)B * S * C * 4);
  float*  scr = (float*) take((size_t)B * T * S * 4);
  (void)take(4096);   // guard pad for pipelined K+64 over-read
  __bf16* rb  = xb;   // x(bf16) dead after q GEMM -> reuse for retrieved
  if (off > ws_size) return;

  const long long nx = (long long)B * T * C;
  const long long nw = (long long)C * C;
  const long long ns = (long long)B * S * C;

  f32_to_bf16_kernel<<<(unsigned)((nx + 255) / 256), 256, 0, stream>>>(x,  xb,  nx);
  f32_to_bf16_kernel<<<(unsigned)((nw + 255) / 256), 256, 0, stream>>>(Wq, wqb, nw);
  f32_to_bf16_kernel<<<(unsigned)((nw + 255) / 256), 256, 0, stream>>>(Wk, wkb, nw);
  f32_to_bf16_kernel<<<(unsigned)((nw + 255) / 256), 256, 0, stream>>>(Wv, wvb, nw);
  f32_to_bf16_kernel<<<(unsigned)((nw + 255) / 256), 256, 0, stream>>>(Wp, wpb, nw);
  pool_slots_kernel<<<(unsigned)((ns + 255) / 256), 256, 0, stream>>>(x, slb, S, C, T / S, T, ns);

  // q = x @ Wq^T  (bf16 out, for scores GEMM)
  gemm_bf16_wmma<<<dim3(C / 64, (B * T) / 32, 1), 32, 0, stream>>>(
      xb, wqb, nullptr, qb, C, C, 0, 0, 0, 1.0f);
  // k = slots @ Wk^T  (bf16 out)
  gemm_bf16_wmma<<<dim3(C / 64, (B * S) / 32, 1), 32, 0, stream>>>(
      slb, wkb, nullptr, kb, C, C, 0, 0, 0, 1.0f);
  // v = slots @ Wv^T  (f32 out, consumed by gather)
  gemm_bf16_wmma<<<dim3(C / 64, (B * S) / 32, 1), 32, 0, stream>>>(
      slb, wvb, vf, nullptr, C, C, 0, 0, 0, 1.0f);
  // scores = (q @ k^T) / sqrt(C), batched over B
  gemm_bf16_wmma<<<dim3(S / 64, T / 32, B), 32, 0, stream>>>(
      qb, kb, scr, nullptr, S, C,
      (long long)T * C, (long long)S * C, (long long)T * S, 0.03125f);
  // top-8 + softmax + weighted gather -> retrieved (bf16)
  topk_gather_kernel<<<B * T, 256, 0, stream>>>(scr, vf, rb, T, S, C);
  // out = 0.5 * retrieved @ Wp^T  (f32 to d_out)
  gemm_bf16_wmma<<<dim3(C / 64, (B * T) / 32, 1), 32, 0, stream>>>(
      rb, wpb, out, nullptr, C, C, 0, 0, 0, 0.5f);
}
